// PALMPartialAttention_2259152798059
// MI455X (gfx1250) — matrix operand order repaired and checked by reading, not compiled, not run on hardware
//
#include <hip/hip_runtime.h>
#include <hip/hip_bf16.h>
#include <math.h>
#include <stdint.h>

// ---------------------------------------------------------------------------
// PALM partial cross-attention block for MI455X (gfx1250, wave32, WMMA).
// GEMMs + attention on v_wmma_f32_16x16x32_f16; A-tiles staged via the
// Tensor Data Mover (tensor_load_to_lds + s_wait_tensorcnt), double-buffered.
// ---------------------------------------------------------------------------

#define B_    2
#define S_    2048
#define H_    2048
#define NH_   16
#define NKV_  4
#define HD_   128
#define G_    4

typedef __attribute__((ext_vector_type(4)))  _Float16 v4h;
typedef __attribute__((ext_vector_type(8)))  _Float16 v8h;
typedef __attribute__((ext_vector_type(16))) _Float16 v16h;
typedef __attribute__((ext_vector_type(8)))  float    v8f;
typedef __attribute__((ext_vector_type(4)))  unsigned int u32x4;
typedef __attribute__((ext_vector_type(8)))  int      i32x8;
typedef __attribute__((ext_vector_type(4)))  int      i32x4;

union hu { v16h h16; v8h h8[2]; };

__device__ __forceinline__ v8f wmma_f16(v16h a, v16h b, v8f c) {
  return __builtin_amdgcn_wmma_f32_16x16x32_f16(false, a, false, b, (short)0, c,
                                                false, false);
}

// Toolchain discriminator for tensor_load_to_lds arity (5-arg ROCm clang-22,
// 6-arg therock clang-23 which ships the TDM header).
#if __has_include(<hip/amd_detail/amd_gfx1250_TDM.h>)
#define TDM_6ARG 1
#endif

// ---------------------------------------------------------------------------
// TDM: DMA a 2D tile of `rows` x 32 halves from a row-major f16 tensor with
// element pitch `stride` into LDS at byte offset lds_off, padding the LDS
// destination by 4 dwords after every 16 dwords -> LDS row pitch = 40 halves.
// Issue once per wave (EXEC ignored by TDM).
// ---------------------------------------------------------------------------
__device__ __forceinline__ void tdm_load_tile_a(const _Float16* gsrc,
                                                unsigned lds_off, int stride,
                                                int dim0, int dim1) {
  u32x4 g0;
  g0[0] = 1u;                                   // count=1, user descriptor
  g0[1] = lds_off;                              // lds_addr (bytes)
  const uint64_t ga = (uint64_t)(uintptr_t)gsrc;
  g0[2] = (unsigned)ga;                         // global_addr[31:0]
  g0[3] = (unsigned)((ga >> 32) & 0x01FFFFFFull) | (2u << 30);  // type=2

  i32x8 g1;
  // data_size=1 (2B), pad_enable, pad_interval=3 (16 dwords), pad_amount=3 (4 dwords)
  g1[0] = (int)((1u << 16) | (1u << 20) | (3u << 22) | (3u << 25));
  const unsigned td0 = (unsigned)dim0, td1 = (unsigned)dim1;
  g1[1] = (int)((td0 & 0xFFFFu) << 16);                 // tensor_dim0 lo16
  g1[2] = (int)((td0 >> 16) | ((td1 & 0xFFFFu) << 16)); // dim0 hi / dim1 lo
  g1[3] = (int)((td1 >> 16) | (32u << 16));             // dim1 hi, tile_dim0=32
  g1[4] = (int)128u;                                    // tile_dim1=128
  g1[5] = (int)(unsigned)stride;                        // tensor_dim0_stride lo32
  g1[6] = 0;                                            // stride hi / dim1_stride lo
  g1[7] = 0;
  i32x4 z4 = {0, 0, 0, 0};
#if defined(TDM_6ARG)
  i32x8 z8 = {0, 0, 0, 0, 0, 0, 0, 0};
  __builtin_amdgcn_tensor_load_to_lds(g0, g1, z4, z4, z8, 0);
#else
  __builtin_amdgcn_tensor_load_to_lds(g0, g1, z4, z4, 0);
#endif
}

// ---------------------------------------------------------------------------
// f32 -> f16 conversion (grid-stride)
// ---------------------------------------------------------------------------
__global__ void cvt_f32_f16(const float* __restrict__ x, _Float16* __restrict__ y,
                            size_t n) {
  size_t i = (size_t)blockIdx.x * blockDim.x + threadIdx.x;
  size_t stride = (size_t)gridDim.x * blockDim.x;
  for (; i < n; i += stride) y[i] = (_Float16)x[i];
}

// ---------------------------------------------------------------------------
// Tiled WMMA GEMM:  Out = epilogue(A[M,K] @ B[K,N] + bias)
// 256 threads (8 waves), tile 128x64, wave tile 32x32. Double-buffered LDS;
// A-tile staged by TDM, B-tile transposed by scalar stores.
// ACT: 1 relu | RES: 0 none, 1 f16, 2 f32 | OUT: 0 f16, 1 f32, 2 f16 V^T
// ---------------------------------------------------------------------------
template <int ACT, int RES, int OUT>
__global__ __launch_bounds__(256) void gemm_f16_k(
    const _Float16* __restrict__ A, const _Float16* __restrict__ Bm,
    const float* __restrict__ bias, const void* __restrict__ Res,
    void* __restrict__ Out, int M, int N, int K, float scale) {
  __shared__ __align__(16) _Float16 a_lds[2][128 * 40];  // [row][k], pitch 40
  __shared__ __align__(16) _Float16 b_lds[2][64 * 40];   // [n][k], pitch 40

  const int tid  = threadIdx.x;
  const int lane = tid & 31;
  const int wid  = tid >> 5;
  const int wm   = wid >> 1;
  const int wn   = wid & 1;
  const int r    = lane & 15;
  const int hi   = lane >> 4;
  const int m0   = blockIdx.y * 128;
  const int n0   = blockIdx.x * 64;
  const int abase = hi ? 8 : 0;
  const int kbase = hi ? 16 : 0;

  v8f acc[2][2] = {};

  auto stage_b = [&](int buf, int k0) {
#pragma unroll
    for (int i = 0; i < 8; ++i) {
      int id = tid + i * 256;
      int k  = id >> 6;
      int n  = id & 63;
      b_lds[buf][n * 40 + k] = Bm[(size_t)(k0 + k) * N + n0 + n];
    }
  };

  // ---- prologue: fill buffer 0 ----
  if (wid == 0)
    tdm_load_tile_a(&A[(size_t)m0 * K], (unsigned)(uintptr_t)&a_lds[0][0], K, K, M);
  stage_b(0, 0);
  if (wid == 0) __builtin_amdgcn_s_wait_tensorcnt(0);
  __syncthreads();

  for (int k0 = 0; k0 < K; k0 += 32) {
    const int buf = (k0 >> 5) & 1;
    const int nk  = k0 + 32;
    // ---- stage next buffer (TDM copy overlaps this iteration's WMMAs) ----
    if (nk < K) {
      if (wid == 0)
        tdm_load_tile_a(&A[(size_t)m0 * K + nk],
                        (unsigned)(uintptr_t)&a_lds[buf ^ 1][0], K, K, M);
      stage_b(buf ^ 1, nk);
      __builtin_prefetch(&Bm[(size_t)(nk + 32 + (tid >> 6)) * N + n0 + (tid & 63)],
                         0, 1);
    }

    // ---- fragments (ISA 16-bit A/B layouts, 16B LDS loads) ----
    hu a0, a1, b0, b1;
    {
      const _Float16* ap = &a_lds[buf][(wm * 32 + r) * 40];
      a0.h8[0] = *(const v8h*)(ap + abase);
      a0.h8[1] = *(const v8h*)(ap + 16 + abase);
      ap += 16 * 40;
      a1.h8[0] = *(const v8h*)(ap + abase);
      a1.h8[1] = *(const v8h*)(ap + 16 + abase);
      const _Float16* bp = &b_lds[buf][(wn * 32 + r) * 40 + abase];
      b0.h8[0] = *(const v8h*)(bp);
      b0.h8[1] = *(const v8h*)(bp + 16);
      bp += 16 * 40;
      b1.h8[0] = *(const v8h*)(bp);
      b1.h8[1] = *(const v8h*)(bp + 16);
    }
    acc[0][0] = wmma_f16(a0.h16, b0.h16, acc[0][0]);
    acc[0][1] = wmma_f16(a0.h16, b1.h16, acc[0][1]);
    acc[1][0] = wmma_f16(a1.h16, b0.h16, acc[1][0]);
    acc[1][1] = wmma_f16(a1.h16, b1.h16, acc[1][1]);

    if (nk < K && wid == 0) __builtin_amdgcn_s_wait_tensorcnt(0);
    __syncthreads();
  }

  // ---- epilogue (C layout: VGPR v -> row v + 8*hi, lane -> col) ----
#pragma unroll
  for (int mt = 0; mt < 2; ++mt) {
#pragma unroll
    for (int nt = 0; nt < 2; ++nt) {
      const int gn  = n0 + wn * 32 + nt * 16 + r;
      const int gmb = m0 + wm * 32 + mt * 16 + hi * 8;
      const float bv = bias[gn];
      v8f c = acc[mt][nt];
#pragma unroll
      for (int v = 0; v < 8; ++v) {
        const int gm = gmb + v;
        float x = c[v] + bv;
        if (ACT) x = fmaxf(x, 0.0f);
        x *= scale;
        if (RES == 1)
          x += (float)((const _Float16*)Res)[(size_t)gm * N + gn];
        if (RES == 2) x += ((const float*)Res)[(size_t)gm * N + gn];
        if (OUT == 0) {
          ((_Float16*)Out)[(size_t)gm * N + gn] = (_Float16)x;
        } else if (OUT == 1) {
          ((float*)Out)[(size_t)gm * N + gn] = x;
        } else {  // transposed V store: [b, kv, d, s]
          const int kvh = gn >> 7;
          const int d   = gn & (HD_ - 1);
          const int bb  = gm >> 11;
          const int s   = gm & (S_ - 1);
          ((_Float16*)Out)[((size_t)(bb * NKV_ + kvh) * HD_ + d) * S_ + s] =
              (_Float16)x;
        }
      }
    }
  }
}

// ---------------------------------------------------------------------------
// Flash attention: one wave per (head, 16-query block). Q in registers as
// 4 A-frags; K/V^T B-frags batch-loaded (one clause, one wait, 8 WMMAs).
// ---------------------------------------------------------------------------
__global__ __launch_bounds__(32) void attn_k(const _Float16* __restrict__ Q,
                                             const _Float16* __restrict__ K,
                                             const _Float16* __restrict__ Vt,
                                             _Float16* __restrict__ ctx) {
  __shared__ __align__(16) _Float16 p_lds[16 * 40];

  const int lane = threadIdx.x;
  const int r    = lane & 15;
  const int hi   = lane >> 4;
  const int abase = hi ? 8 : 0;
  const int kbase = hi ? 16 : 0;

  const int head = blockIdx.x;
  const int b    = head / NH_;
  const int qh   = head % NH_;
  const int kv   = qh / G_;
  const int q0   = blockIdx.y * 16;

  const _Float16* Qp = Q + (size_t)(b * S_ + q0) * H_ + (size_t)qh * HD_;
  const _Float16* Kp = K + (size_t)b * S_ * (NKV_ * HD_) + (size_t)kv * HD_;
  const _Float16* Vp = Vt + (size_t)(b * NKV_ + kv) * HD_ * S_;
  _Float16*       Cp = ctx + (size_t)(b * S_ + q0) * H_ + (size_t)qh * HD_;

  v16h qf[4];
#pragma unroll
  for (int kc = 0; kc < 4; ++kc) {
    const _Float16* src = Qp + (size_t)r * H_ + kc * 32;
    hu u;
    u.h8[0] = *(const v8h*)(src + abase);
    u.h8[1] = *(const v8h*)(src + 16 + abase);
    qf[kc] = u.h16;
  }

  float m_i[8], l_i[8];
  v8f acc[8] = {};
#pragma unroll
  for (int v = 0; v < 8; ++v) {
    m_i[v] = -__builtin_inff();
    l_i[v] = 0.0f;
  }

  for (int s0 = 0; s0 < S_; s0 += 32) {
    // ---- batch-load all K fragments, then 8 back-to-back WMMAs ----
    hu kb[4][2];
#pragma unroll
    for (int kc = 0; kc < 4; ++kc)
#pragma unroll
      for (int nt = 0; nt < 2; ++nt) {
        const _Float16* kp =
            Kp + (size_t)(s0 + nt * 16 + r) * (NKV_ * HD_) + kc * 32 + abase;
        kb[kc][nt].h8[0] = *(const v8h*)kp;
        kb[kc][nt].h8[1] = *(const v8h*)(kp + 16);
      }
    v8f sc[2] = {};
#pragma unroll
    for (int kc = 0; kc < 4; ++kc) {
      sc[0] = wmma_f16(qf[kc], kb[kc][0].h16, sc[0]);
      sc[1] = wmma_f16(qf[kc], kb[kc][1].h16, sc[1]);
    }

    // ---- start V-fragment loads; they fly under the softmax math ----
    hu vb[8];
#pragma unroll
    for (int t = 0; t < 8; ++t) {
      const _Float16* vp = Vp + (size_t)(t * 16 + r) * S_ + s0 + abase;
      vb[t].h8[0] = *(const v8h*)vp;
      vb[t].h8[1] = *(const v8h*)(vp + 16);
    }

    // ---- online softmax row statistics (C-layout row v + 8*hi) ----
    float alpha[8];
#pragma unroll
    for (int v = 0; v < 8; ++v) {
      float mx = fmaxf(sc[0][v], sc[1][v]);
#pragma unroll
      for (int m = 8; m >= 1; m >>= 1) mx = fmaxf(mx, __shfl_xor(mx, m, 32));
      const float mn = fmaxf(m_i[v], mx);
      alpha[v] = __expf(m_i[v] - mn);
      const float p0 = __expf(sc[0][v] - mn);
      const float p1 = __expf(sc[1][v] - mn);
      float rs = p0 + p1;
#pragma unroll
      for (int m = 8; m >= 1; m >>= 1) rs += __shfl_xor(rs, m, 32);
      l_i[v] = l_i[v] * alpha[v] + rs;
      m_i[v] = mn;
      p_lds[(v + 8 * hi) * 40 + r]      = (_Float16)p0;
      p_lds[(v + 8 * hi) * 40 + 16 + r] = (_Float16)p1;
    }
    __syncthreads();

    // ---- probs C-layout -> A-fragment ----
    hu pu;
    {
      const _Float16* pp = &p_lds[(size_t)r * 40];
      pu.h8[0] = *(const v8h*)(pp + abase);
      pu.h8[1] = *(const v8h*)(pp + 16 + abase);
    }
    __syncthreads();

    // ---- rescale accumulators, PV: ctx += P(16x32) @ V(32x128) ----
#pragma unroll
    for (int t = 0; t < 8; ++t) {
      v8f a = acc[t];
#pragma unroll
      for (int v = 0; v < 8; ++v) a[v] *= alpha[v];
      acc[t] = wmma_f16(pu.h16, vb[t].h16, a);
    }
  }

  // ---- normalize and store context ----
#pragma unroll
  for (int t = 0; t < 8; ++t) {
#pragma unroll
    for (int v = 0; v < 8; ++v) {
      const float o = acc[t][v] / l_i[v];
      Cp[(size_t)(v + 8 * hi) * H_ + t * 16 + r] = (_Float16)o;
    }
  }
}

// ---------------------------------------------------------------------------
// Row LayerNorm over H_=2048, one 256-thread block per row.
// ---------------------------------------------------------------------------
__global__ __launch_bounds__(256) void layernorm_k(const float* __restrict__ x,
                                                   const float* __restrict__ gamma,
                                                   const float* __restrict__ beta,
                                                   float* __restrict__ y) {
  __shared__ float s1[256], s2[256];
  const int row = blockIdx.x;
  const int tid = threadIdx.x;
  const float* xr = x + (size_t)row * H_;
  float a = 0.0f, b = 0.0f;
  for (int i = tid; i < H_; i += 256) {
    const float v = xr[i];
    a += v;
    b += v * v;
  }
  s1[tid] = a;
  s2[tid] = b;
  __syncthreads();
  for (int s = 128; s > 0; s >>= 1) {
    if (tid < s) {
      s1[tid] += s1[tid + s];
      s2[tid] += s2[tid + s];
    }
    __syncthreads();
  }
  const float mu  = s1[0] * (1.0f / H_);
  const float var = s2[0] * (1.0f / H_) - mu * mu;
  const float inv = rsqrtf(var + 1e-12f);
  for (int i = tid; i < H_; i += 256)
    y[(size_t)row * H_ + i] = (xr[i] - mu) * inv * gamma[i] + beta[i];
}

// ---------------------------------------------------------------------------
// Host launcher
// ---------------------------------------------------------------------------
extern "C" void kernel_launch(void* const* d_in, const int* in_sizes, int n_in,
                              void* d_out, int out_size, void* d_ws,
                              size_t ws_size, hipStream_t stream) {
  (void)in_sizes; (void)n_in; (void)out_size; (void)ws_size;

  const float* hidden = (const float*)d_in[0];
  const float* source = (const float*)d_in[1];
  const float* Wq = (const float*)d_in[2];
  const float* bq = (const float*)d_in[3];
  const float* Wk = (const float*)d_in[4];
  const float* bk = (const float*)d_in[5];
  const float* Wv = (const float*)d_in[6];
  const float* bv = (const float*)d_in[7];
  const float* Wd = (const float*)d_in[8];
  const float* bd = (const float*)d_in[9];
  const float* W1 = (const float*)d_in[10];
  const float* b1 = (const float*)d_in[11];
  const float* W2 = (const float*)d_in[12];
  const float* b2 = (const float*)d_in[13];
  const float* gamma = (const float*)d_in[14];
  const float* beta  = (const float*)d_in[15];

  const size_t BSH  = (size_t)B_ * S_ * H_;
  const size_t HH   = (size_t)H_ * H_;
  const size_t HKV  = (size_t)H_ * (NKV_ * HD_);
  const size_t BSKV = (size_t)B_ * S_ * (NKV_ * HD_);

  char* w = (char*)d_ws;
  size_t off = 0;
  auto take = [&](size_t elems, size_t esz) -> void* {
    void* p = w + off;
    off += (elems * esz + 255) & ~(size_t)255;
    return p;
  };
  _Float16* hs16  = (_Float16*)take(BSH, 2);
  _Float16* src16 = (_Float16*)take(BSH, 2);
  _Float16* W1h   = (_Float16*)take(HH, 2);
  _Float16* W2h   = (_Float16*)take(HH, 2);
  _Float16* Wqh   = (_Float16*)take(HH, 2);
  _Float16* Wkh   = (_Float16*)take(HKV, 2);
  _Float16* Wvh   = (_Float16*)take(HKV, 2);
  _Float16* Wdh   = (_Float16*)take(HH, 2);
  _Float16* Pl1h  = (_Float16*)take(BSH, 2);
  _Float16* Ph    = (_Float16*)take(BSH, 2);
  _Float16* Qh    = (_Float16*)take(BSH, 2);
  _Float16* Kh    = (_Float16*)take(BSKV, 2);
  _Float16* Vth   = (_Float16*)take(BSKV, 2);
  _Float16* Ch    = (_Float16*)take(BSH, 2);
  float*    attnf = (float*)take(BSH, 4);

  cvt_f32_f16<<<1024, 256, 0, stream>>>(hidden, hs16, BSH);
  cvt_f32_f16<<<1024, 256, 0, stream>>>(source, src16, BSH);
  cvt_f32_f16<<<1024, 256, 0, stream>>>(W1, W1h, HH);
  cvt_f32_f16<<<1024, 256, 0, stream>>>(W2, W2h, HH);
  cvt_f32_f16<<<1024, 256, 0, stream>>>(Wq, Wqh, HH);
  cvt_f32_f16<<<1024, 256, 0, stream>>>(Wk, Wkh, HKV);
  cvt_f32_f16<<<1024, 256, 0, stream>>>(Wv, Wvh, HKV);
  cvt_f32_f16<<<1024, 256, 0, stream>>>(Wd, Wdh, HH);

  const dim3 blk(256);
  const dim3 g2048(2048 / 64, 4096 / 128);
  const dim3 g512(512 / 64, 4096 / 128);
  const float qscale = 0.08838834764831845f;  // 1/sqrt(HD)

  gemm_f16_k<1, 0, 0><<<g2048, blk, 0, stream>>>(src16, W1h, b1, nullptr, Pl1h,
                                                 4096, 2048, 2048, 1.0f);
  gemm_f16_k<0, 1, 0><<<g2048, blk, 0, stream>>>(Pl1h, W2h, b2, Pl1h, Ph, 4096,
                                                 2048, 2048, 1.0f);
  gemm_f16_k<0, 0, 0><<<g2048, blk, 0, stream>>>(hs16, Wqh, bq, nullptr, Qh,
                                                 4096, 2048, 2048, qscale);
  gemm_f16_k<0, 0, 0><<<g512, blk, 0, stream>>>(Ph, Wkh, bk, nullptr, Kh, 4096,
                                                512, 2048, 1.0f);
  gemm_f16_k<0, 0, 2><<<g512, blk, 0, stream>>>(Ph, Wvh, bv, nullptr, Vth, 4096,
                                                512, 2048, 1.0f);
  attn_k<<<dim3(B_ * NH_, S_ / 16), 32, 0, stream>>>(Qh, Kh, Vth, Ch);
  gemm_f16_k<0, 2, 1><<<g2048, blk, 0, stream>>>(Ch, Wdh, bd, hidden, attnf,
                                                 4096, 2048, 2048, 1.0f);
  layernorm_k<<<4096, 256, 0, stream>>>(attnf, gamma, beta, (float*)d_out);
}
